// FeedBack_54966991454249
// MI455X (gfx1250) — compile-verified, matrix operand order checked by reading, not run on hardware
//
#include <hip/hip_runtime.h>

typedef __attribute__((ext_vector_type(16))) __bf16 v16bf;
typedef __attribute__((ext_vector_type(8)))  float  v8f;

// Problem constants
constexpr int kB    = 4096;
constexpr int kTin  = 168;
constexpr int kTout = 24;
constexpr int kF    = 32;
constexpr int kU    = 512;
constexpr int kKW   = kF + kU;      // 544 : concatenated [x | h] K-dim
constexpr int k4U   = 4 * kU;       // 2048

// Tiling
constexpr int kBM = 32;                         // batch rows per workgroup
constexpr int kSA = kF + 2 * kU + 16;           // 1072 : a_buf row stride (bf16), x + double-buffered h + pad
constexpr int kCS = kU + 4;                     // 516  : c_buf row stride (f32), bank-skewed

__device__ __forceinline__ float sigmoidf(float x) { return 1.0f / (1.0f + __expf(-x)); }
__device__ __forceinline__ float siluf(float x)    { return x * sigmoidf(x); }

// ---------------------------------------------------------------------------
// Prep: f32 weights -> bf16, transposed to [N][K] so B-fragments are contiguous
//   Wcat_t[n][k] = (k < 32) ? Wk[k][n] : Wr[k-32][n]     (2048 x 544)
//   Wd_t[f][u]   = Wd[u][f]                              (32 x 512)
// ---------------------------------------------------------------------------
__global__ void prep_weights(const float* __restrict__ Wk,
                             const float* __restrict__ Wr,
                             const float* __restrict__ Wd,
                             __bf16* __restrict__ Wcat_t,
                             __bf16* __restrict__ Wd_t) {
  int idx = blockIdx.x * blockDim.x + threadIdx.x;
  if (idx < k4U * kKW) {
    int n = idx / kKW;
    int k = idx % kKW;
    float v = (k < kF) ? Wk[(size_t)k * k4U + n] : Wr[(size_t)(k - kF) * k4U + n];
    Wcat_t[idx] = (__bf16)v;
  }
  if (idx < kF * kU) {
    int f = idx / kU;
    int u = idx % kU;
    Wd_t[idx] = (__bf16)Wd[(size_t)u * kF + f];
  }
}

// ---------------------------------------------------------------------------
// One LSTM step for a 32-row batch tile.
// Wave w owns units [w*64, w*64+64) -> 4 unit-tiles of 16 (rolled loop: only
// one tile's 64 accumulator VGPRs live at a time -> no spills).
// Reads A = [x | h(p)] from LDS, writes h into region (p^1) (double buffer,
// avoids cross-wave RAW within a step). c is wave-private columns (no hazard).
// ---------------------------------------------------------------------------
__device__ __forceinline__ void lstm_step(__bf16* a_buf, float* c_buf,
                                          const __bf16* __restrict__ Wt,
                                          const float* __restrict__ bias,
                                          int wave, int lane, int p) {
  const int lm = lane & 15;
  const int lh = lane >> 4;
  const int hread  = kF + p * kU;
  const int hwrite = kF + (p ^ 1) * kU;

#pragma unroll 1
  for (int ut = 0; ut < 4; ++ut) {
    const int n0 = wave * 64 + ut * 16;
    const int n  = n0 + lm;

    v8f acc[4][2];
#pragma unroll
    for (int g = 0; g < 4; ++g) {
      acc[g][0] = {};
      acc[g][1] = {};
    }

    // per-lane weight row pointers (B fragment: 16 contiguous K values)
    const __bf16* w0 = Wt + (size_t)(0 * kU + n) * kKW + lh * 16;
    const __bf16* w1 = Wt + (size_t)(1 * kU + n) * kKW + lh * 16;
    const __bf16* w2 = Wt + (size_t)(2 * kU + n) * kKW + lh * 16;
    const __bf16* w3 = Wt + (size_t)(3 * kU + n) * kKW + lh * 16;

#pragma unroll 2
    for (int kk = 0; kk < 17; ++kk) {
      const int cb = (kk == 0) ? 0 : (hread + (kk - 1) * 32);  // a_buf column base
      const int ko = kk * 32;                                  // weight K base

      v16bf a0 = *(const v16bf*)&a_buf[(size_t)lm        * kSA + cb + lh * 16];
      v16bf a1 = *(const v16bf*)&a_buf[(size_t)(lm + 16) * kSA + cb + lh * 16];

      v16bf b0 = *(const v16bf*)(w0 + ko);
      v16bf b1 = *(const v16bf*)(w1 + ko);
      v16bf b2 = *(const v16bf*)(w2 + ko);
      v16bf b3 = *(const v16bf*)(w3 + ko);

      acc[0][0] = __builtin_amdgcn_wmma_f32_16x16x32_bf16(false, a0, false, b0, (short)0, acc[0][0], false, false);
      acc[0][1] = __builtin_amdgcn_wmma_f32_16x16x32_bf16(false, a1, false, b0, (short)0, acc[0][1], false, false);
      acc[1][0] = __builtin_amdgcn_wmma_f32_16x16x32_bf16(false, a0, false, b1, (short)0, acc[1][0], false, false);
      acc[1][1] = __builtin_amdgcn_wmma_f32_16x16x32_bf16(false, a1, false, b1, (short)0, acc[1][1], false, false);
      acc[2][0] = __builtin_amdgcn_wmma_f32_16x16x32_bf16(false, a0, false, b2, (short)0, acc[2][0], false, false);
      acc[2][1] = __builtin_amdgcn_wmma_f32_16x16x32_bf16(false, a1, false, b2, (short)0, acc[2][1], false, false);
      acc[3][0] = __builtin_amdgcn_wmma_f32_16x16x32_bf16(false, a0, false, b3, (short)0, acc[3][0], false, false);
      acc[3][1] = __builtin_amdgcn_wmma_f32_16x16x32_bf16(false, a1, false, b3, (short)0, acc[3][1], false, false);
    }

    const float bi = bias[0 * kU + n];
    const float bf = bias[1 * kU + n];
    const float bg = bias[2 * kU + n];
    const float bo = bias[3 * kU + n];

#pragma unroll
    for (int rt = 0; rt < 2; ++rt) {
#pragma unroll
      for (int v = 0; v < 8; ++v) {
        const int m = v + 8 * lh + rt * 16;  // C-fragment row for this vgpr/lane
        const float zi = acc[0][rt][v] + bi;
        const float zf = acc[1][rt][v] + bf;
        const float zg = acc[2][rt][v] + bg;
        const float zo = acc[3][rt][v] + bo;
        const float cold = c_buf[(size_t)m * kCS + n];
        const float cn = sigmoidf(zf) * cold + sigmoidf(zi) * siluf(zg);
        c_buf[(size_t)m * kCS + n] = cn;
        const float h = sigmoidf(zo) * siluf(cn);
        a_buf[(size_t)m * kSA + hwrite + n] = (__bf16)h;
      }
    }
  }
}

// ---------------------------------------------------------------------------
// Dense projection pred = h @ Wd + bd  (32x512 x 512x32).
// Waves 0..3 each compute one 16x16 tile. Writes f32 to d_out and bf16 back
// into the x-region of a_buf (next decode input).
// ---------------------------------------------------------------------------
__device__ __forceinline__ void dense_step(__bf16* a_buf,
                                           const __bf16* __restrict__ Wdt,
                                           const float* __restrict__ bd,
                                           float* __restrict__ out,
                                           int b0, int step, int wave, int lane, int p) {
  if (wave < 4) {
    const int lm = lane & 15;
    const int lh = lane >> 4;
    const int rt = wave >> 1;
    const int n  = (wave & 1) * 16 + lm;
    const int hread = kF + p * kU;

    v8f acc = {};
    const __bf16* wg = Wdt + (size_t)n * kU + lh * 16;
#pragma unroll 2
    for (int kk = 0; kk < 16; ++kk) {
      v16bf a  = *(const v16bf*)&a_buf[(size_t)(lm + rt * 16) * kSA + hread + kk * 32 + lh * 16];
      v16bf bw = *(const v16bf*)(wg + kk * 32);
      acc = __builtin_amdgcn_wmma_f32_16x16x32_bf16(false, a, false, bw, (short)0, acc, false, false);
    }

    const float bdv = bd[n];
#pragma unroll
    for (int v = 0; v < 8; ++v) {
      const int m = v + 8 * lh + rt * 16;
      const float pv = acc[v] + bdv;
      out[(size_t)(b0 + m) * (kTout * kF) + (size_t)step * kF + n] = pv;
      a_buf[(size_t)m * kSA + n] = (__bf16)pv;  // feed back as next input
    }
  }
}

// ---------------------------------------------------------------------------
// Main kernel: one workgroup (8 waves) owns 32 batch rows through all steps.
// ---------------------------------------------------------------------------
__global__ __launch_bounds__(256, 1)
void feedback_lstm(const float* __restrict__ x,
                   const float* __restrict__ bias,
                   const float* __restrict__ bd,
                   const __bf16* __restrict__ Wt,
                   const __bf16* __restrict__ Wdt,
                   float* __restrict__ out) {
  __shared__ __attribute__((aligned(32))) __bf16 a_buf[kBM * kSA];
  __shared__ __attribute__((aligned(16))) float  c_buf[kBM * kCS];

  const int tid  = threadIdx.x;
  const int wave = tid >> 5;
  const int lane = tid & 31;
  const int b0   = blockIdx.x * kBM;

  for (int i = tid; i < kBM * kSA; i += 256) a_buf[i] = (__bf16)0.0f;
  for (int i = tid; i < kBM * kCS; i += 256) c_buf[i] = 0.0f;
  __syncthreads();

  int p = 0;

  // ---- warmup over input history ----
  const int xm = tid >> 3;        // 0..31 row
  const int xc = (tid & 7) * 4;   // 0..28 col, 4 floats each
#pragma unroll 1
  for (int t = 0; t < kTin; ++t) {
    const float4 xv = *(const float4*)&x[(size_t)(b0 + xm) * (kTin * kF) + (size_t)t * kF + xc];
    a_buf[(size_t)xm * kSA + xc + 0] = (__bf16)xv.x;
    a_buf[(size_t)xm * kSA + xc + 1] = (__bf16)xv.y;
    a_buf[(size_t)xm * kSA + xc + 2] = (__bf16)xv.z;
    a_buf[(size_t)xm * kSA + xc + 3] = (__bf16)xv.w;
    __syncthreads();
    lstm_step(a_buf, c_buf, Wt, bias, wave, lane, p);
    p ^= 1;
    __syncthreads();
  }

  // ---- first prediction ----
  dense_step(a_buf, Wdt, bd, out, b0, 0, wave, lane, p);
  __syncthreads();

  // ---- autoregressive decode ----
#pragma unroll 1
  for (int s = 1; s < kTout; ++s) {
    lstm_step(a_buf, c_buf, Wt, bias, wave, lane, p);
    p ^= 1;
    __syncthreads();
    dense_step(a_buf, Wdt, bd, out, b0, s, wave, lane, p);
    __syncthreads();
  }
}

// ---------------------------------------------------------------------------
extern "C" void kernel_launch(void* const* d_in, const int* in_sizes, int n_in,
                              void* d_out, int out_size, void* d_ws, size_t ws_size,
                              hipStream_t stream) {
  const float* x    = (const float*)d_in[0];
  const float* Wk   = (const float*)d_in[1];
  const float* Wr   = (const float*)d_in[2];
  const float* bias = (const float*)d_in[3];
  const float* Wd   = (const float*)d_in[4];
  const float* bd   = (const float*)d_in[5];
  float* out = (float*)d_out;

  __bf16* Wt  = (__bf16*)d_ws;                                        // 2048*544 bf16
  __bf16* Wdt = (__bf16*)((char*)d_ws + (size_t)k4U * kKW * 2);       // 32*512 bf16

  const int prep_elems = k4U * kKW;
  prep_weights<<<(prep_elems + 255) / 256, 256, 0, stream>>>(Wk, Wr, Wd, Wt, Wdt);

  feedback_lstm<<<kB / kBM, 256, 0, stream>>>(x, bias, bd, Wt, Wdt, out);
}
